// BRAR_1992864825632
// MI455X (gfx1250) — compile-verified, matrix operand order checked by reading, not run on hardware
//
#include <hip/hip_runtime.h>
#include <cstdint>
#include <cstddef>

typedef __attribute__((ext_vector_type(16))) __bf16   v16bf;
typedef __attribute__((ext_vector_type(8)))  float    v8f;
typedef __attribute__((ext_vector_type(8)))  unsigned v8u;
typedef __attribute__((ext_vector_type(4)))  unsigned u32x4;
typedef __attribute__((ext_vector_type(8)))  int      i32x8;
typedef __attribute__((ext_vector_type(4)))  int      i32x4;

#define SLEN   4096
#define EDIM   300
#define EPAD   320
#define HDIM   256       // HD
#define NGATE  1024      // 4*HD
#define KTAG   6
#define KT_IH  10        // EPAD/32
#define NT_IH  128       // 2048/16
#define MTILES 256       // SLEN/16
#define KT_HH  8         // 256/32
#define NT_HH  64        // 1024/16
#define NEGV   -10000.0f

#if defined(__has_builtin)
#if __has_builtin(__builtin_amdgcn_tensor_load_to_lds)
#define USE_TDM 1
#endif
#endif
#ifndef USE_TDM
#define USE_TDM 0
#endif

__device__ __forceinline__ unsigned short f2bf(float f) {
  unsigned u = __builtin_bit_cast(unsigned, f);
  unsigned r = u + 0x7fffu + ((u >> 16) & 1u);
  return (unsigned short)(r >> 16);
}

// K index for half j (0..15) of the 16-bit WMMA operand register layout.
__device__ __forceinline__ int kpat(int lane, int j) {
  int p = j >> 1;
  return ((p & 4) ? 16 : 0) + ((lane & 16) ? 8 : 0) + ((p & 3) << 1) + (j & 1);
}

// ---------------- pack kernels: build WMMA-layout bf16 tiles ----------------

__global__ void pack_a(const int* __restrict__ sent, const float* __restrict__ emb,
                       unsigned short* __restrict__ Ap) {
  int tid = blockIdx.x * blockDim.x + threadIdx.x;
  if (tid >= MTILES * KT_IH * 32) return;
  int tile = tid >> 5, l = tid & 31;
  int mt = tile / KT_IH, kt = tile % KT_IH;
  int s = mt * 16 + (l & 15);
  size_t row = (size_t)sent[s];
  unsigned short* out = Ap + (size_t)tile * 512 + l * 16;
#pragma unroll
  for (int j = 0; j < 16; ++j) {
    int k = kt * 32 + kpat(l, j);
    float v = (k < EDIM) ? emb[row * EDIM + k] : 0.f;
    out[j] = f2bf(v);
  }
}

__global__ void pack_bih(const float* __restrict__ Wf, const float* __restrict__ Wb,
                         unsigned short* __restrict__ Bp) {
  int tid = blockIdx.x * blockDim.x + threadIdx.x;
  if (tid >= KT_IH * NT_IH * 32) return;
  int tile = tid >> 5, l = tid & 31;
  int kt = tile / NT_IH, nt = tile % NT_IH;
  int n = nt * 16 + (l & 15);
  const float* W = (n < NGATE) ? Wf : Wb;
  int nl = n & (NGATE - 1);
  unsigned short* out = Bp + (size_t)tile * 512 + l * 16;
#pragma unroll
  for (int j = 0; j < 16; ++j) {
    int k = kt * 32 + kpat(l, j);
    float v = (k < EDIM) ? W[(size_t)nl * EDIM + k] : 0.f;
    out[j] = f2bf(v);
  }
}

__global__ void pack_bhh(const float* __restrict__ Wf, const float* __restrict__ Wb,
                         unsigned short* __restrict__ Bhh) {
  int tid = blockIdx.x * blockDim.x + threadIdx.x;
  if (tid >= 2 * KT_HH * NT_HH * 32) return;
  int tile = tid >> 5, l = tid & 31;
  int dir = tile >> 9, rem = tile & 511;
  int kt = rem / NT_HH, nt = rem % NT_HH;
  int n = nt * 16 + (l & 15);
  const float* W = dir ? Wb : Wf;
  unsigned short* out = Bhh + (size_t)tile * 512 + l * 16;
#pragma unroll
  for (int j = 0; j < 16; ++j) {
    int k = kt * 32 + kpat(l, j);
    out[j] = f2bf(W[(size_t)n * HDIM + k]);
  }
}

// ---------------- GEMM: xW[4096,2048] = x[4096,320] * [W_ih_f|W_ih_b]^T + bias ----

__global__ __launch_bounds__(256) void gemm_xw(const unsigned short* __restrict__ Ap,
                                               const unsigned short* __restrict__ Bp,
                                               const float* __restrict__ b_f,
                                               const float* __restrict__ b_b,
                                               float* __restrict__ xW) {
  int wid = threadIdx.x >> 5, l = threadIdx.x & 31;
  int gw = blockIdx.x * 8 + wid;          // 8192 waves total
  int mt = gw >> 5, ng = gw & 31;         // wave computes rows mt*16.., cols ng*64..
  const v8u* A8 = (const v8u*)Ap;
  const v8u* B8 = (const v8u*)Bp;
  v8f acc[4] = {v8f{}, v8f{}, v8f{}, v8f{}};
  for (int kt = 0; kt < KT_IH; ++kt) {
    v16bf a = __builtin_bit_cast(v16bf, A8[((size_t)(mt * KT_IH + kt)) * 32 + l]);
    if (kt + 1 < KT_IH)
      __builtin_prefetch(&B8[((size_t)((kt + 1) * NT_IH + ng * 4)) * 32 + l], 0, 0);
#pragma unroll
    for (int q = 0; q < 4; ++q) {
      int nt = ng * 4 + q;
      v16bf b = __builtin_bit_cast(v16bf, B8[((size_t)(kt * NT_IH + nt)) * 32 + l]);
      acc[q] = __builtin_amdgcn_wmma_f32_16x16x32_bf16(false, a, false, b,
                                                       (short)0, acc[q], false, false);
    }
  }
  int rbase = mt * 16 + ((l & 16) ? 8 : 0);
#pragma unroll
  for (int q = 0; q < 4; ++q) {
    int n = ng * 64 + q * 16 + (l & 15);
    float bias = (n < NGATE) ? b_f[n] : b_b[n - NGATE];
#pragma unroll
    for (int r = 0; r < 8; ++r)
      xW[(size_t)(rbase + r) * 2048 + n] = acc[q][r] + bias;
  }
}

// ---------------- LSTM scan: recurrent matvec on the matrix pipe ----------------
// blockIdx.x = direction (0 fwd, 1 bwd), 1024 threads (32 waves).
// Per step: h broadcast into all 16 A-rows; 32 waves x (2 N-tiles x 8 K-chunks) of
// V_WMMA_F32_16X16X32_BF16 produce the 1024 gate pre-activations. The xW row for
// step t+1 is DMA'd into a double-buffered LDS slab by the Tensor Data Mover so
// phase B never stalls on L2 (s_wait_tensorcnt <= 1 per step).

__global__ __launch_bounds__(1024) void lstm_k(const unsigned short* __restrict__ Bhh,
                                               const float* __restrict__ xW,
                                               const float* __restrict__ h0,
                                               const float* __restrict__ c0,
                                               float* __restrict__ lstm_out,
                                               float* __restrict__ final_vec) {
  __shared__ __align__(16) unsigned short h_lds[HDIM];
  __shared__ float gates[NGATE];
#if USE_TDM
  __shared__ __align__(16) float xwbuf[2][NGATE];
#endif
  int dir = blockIdx.x;
  int tid = threadIdx.x;
  float c = 0.f;
  if (tid < HDIM) {
    c = c0[dir * HDIM + tid];
    h_lds[tid] = f2bf(h0[dir * HDIM + tid]);
  }
  int wid = tid >> 5, l = tid & 31;
  int nt0 = wid * 2;
  const v8u* B8 = (const v8u*)Bhh;

#if USE_TDM
  // TDM descriptor: 1-D tile of 1024 f32 elements, global -> LDS.
  auto issue_row = [&](int rowidx, int buf) {
    size_t ga = (size_t)(const void*)(xW + (size_t)rowidx * 2048 + dir * NGATE);
    unsigned lds = (unsigned)(uintptr_t)(&xwbuf[buf][0]);
    u32x4 g0;
    g0[0] = 1u;                                            // count=1, user mode
    g0[1] = lds;                                           // lds_addr [63:32]
    g0[2] = (unsigned)(ga & 0xffffffffu);                  // global_addr lo
    g0[3] = (unsigned)((ga >> 32) & 0x01ffffffu) | 0x80000000u; // ga hi | type=2
    i32x8 g1;
    g1[0] = (int)(2u << 16);       // workgroup_mask=0, data_size=2 (4 bytes)
    g1[1] = (int)(1024u << 16);    // tensor_dim0 = 1024 (low 16 bits @ bit 48)
    g1[2] = (int)(1u << 16);       // tensor_dim0 hi = 0, tensor_dim1 = 1
    g1[3] = (int)(1024u << 16);    // tensor_dim1 hi = 0, tile_dim0 = 1024
    g1[4] = 1;                     // tile_dim1 = 1, tile_dim2 = 0
    g1[5] = 1024;                  // tensor_dim0_stride lo
    g1[6] = 0;                     // stride hi, tensor_dim1_stride lo
    g1[7] = 0;
    __builtin_amdgcn_tensor_load_to_lds(g0, g1, (i32x4)0, (i32x4)0, (i32x8)0, 0);
  };
  if (wid == 0) issue_row(dir ? (SLEN - 1) : 0, 0);
#endif
  __syncthreads();

  for (int t = 0; t < SLEN; ++t) {
    int row = dir ? (SLEN - 1 - t) : t;
#if USE_TDM
    if (wid == 0 && t + 1 < SLEN) {
      int nrow = dir ? (SLEN - 2 - t) : (t + 1);
      issue_row(nrow, (t + 1) & 1);
    }
#endif
    v8f acc0 = v8f{}, acc1 = v8f{};
    const uint4* hv = (const uint4*)h_lds;
    int hbase = ((l & 16) ? 1 : 0);
#pragma unroll
    for (int kc = 0; kc < KT_HH; ++kc) {
      // A operand: pairs for VGPRs 0-3 and 4-7 are each 16 contiguous bytes of h.
      uint4 q0 = hv[kc * 4 + hbase];        // K = kc*32 + base .. +7
      uint4 q1 = hv[kc * 4 + hbase + 2];    // K = kc*32 + 16 + base .. +7
      v8u av;
      av[0] = q0.x; av[1] = q0.y; av[2] = q0.z; av[3] = q0.w;
      av[4] = q1.x; av[5] = q1.y; av[6] = q1.z; av[7] = q1.w;
      v16bf a = __builtin_bit_cast(v16bf, av);
      v16bf b0 = __builtin_bit_cast(
          v16bf, B8[((size_t)((dir * KT_HH + kc) * NT_HH + nt0)) * 32 + l]);
      v16bf b1 = __builtin_bit_cast(
          v16bf, B8[((size_t)((dir * KT_HH + kc) * NT_HH + nt0 + 1)) * 32 + l]);
      acc0 = __builtin_amdgcn_wmma_f32_16x16x32_bf16(false, a, false, b0,
                                                     (short)0, acc0, false, false);
      acc1 = __builtin_amdgcn_wmma_f32_16x16x32_bf16(false, a, false, b1,
                                                     (short)0, acc1, false, false);
    }
    // all A-rows identical -> lane's element 0 is gate(n = tile*16 + (l&15))
    if (l < 16) {
      gates[nt0 * 16 + l] = acc0[0];
      gates[(nt0 + 1) * 16 + l] = acc1[0];
    }
#if USE_TDM
    if (wid == 0) {
      if (t + 1 < SLEN) __builtin_amdgcn_s_wait_tensorcnt(1); // row t done
      else              __builtin_amdgcn_s_wait_tensorcnt(0);
    }
#endif
    __syncthreads();
    if (tid < HDIM) {
#if USE_TDM
      const float* xwr = xwbuf[t & 1];
#else
      const float* xwr = xW + (size_t)row * 2048 + dir * NGATE;
#endif
      float ig = gates[tid] + xwr[tid];
      float fg = gates[HDIM + tid] + xwr[HDIM + tid];
      float gg = gates[2 * HDIM + tid] + xwr[2 * HDIM + tid];
      float og = gates[3 * HDIM + tid] + xwr[3 * HDIM + tid];
      float si = 1.f / (1.f + __expf(-ig));
      float sf = 1.f / (1.f + __expf(-fg));
      float so = 1.f / (1.f + __expf(-og));
      c = sf * c + si * tanhf(gg);
      float h = so * tanhf(c);
      h_lds[tid] = f2bf(h);
      lstm_out[(size_t)row * 512 + dir * HDIM + tid] = h;
      if (t == SLEN - 1) final_vec[dir * HDIM + tid] = h;
    }
    __syncthreads();
  }
}

// ---------------- attention: softmax(lstm_out @ final) folded into per-tag bias ----

__global__ __launch_bounds__(1024) void attn_k(const float* __restrict__ lstm_out,
                                               const float* __restrict__ final_vec,
                                               const float* __restrict__ attn_w,
                                               const float* __restrict__ W_tag,
                                               const float* __restrict__ b_tag,
                                               float* __restrict__ bias2) {
  __shared__ float fsh[512];
  __shared__ float sc[SLEN];
  __shared__ float red[1024];
  __shared__ float ao[512];
  int tid = threadIdx.x;
  if (tid < 512) fsh[tid] = final_vec[tid];
  __syncthreads();
  int wid = tid >> 5, l = tid & 31;
  for (int i = 0; i < 128; ++i) {           // 32 waves x 128 rows
    int t = wid * 128 + i;
    float p = 0.f;
    for (int h = l; h < 512; h += 32) p += lstm_out[(size_t)t * 512 + h] * fsh[h];
#pragma unroll
    for (int off = 16; off; off >>= 1) p += __shfl_xor(p, off, 32);
    if (l == 0) sc[t] = p;
  }
  __syncthreads();
  float m = -3.4e38f;
  for (int i = tid; i < SLEN; i += 1024) m = fmaxf(m, sc[i]);
  red[tid] = m; __syncthreads();
  for (int st = 512; st; st >>= 1) { if (tid < st) red[tid] = fmaxf(red[tid], red[tid + st]); __syncthreads(); }
  m = red[0]; __syncthreads();
  float ss = 0.f;
  for (int i = tid; i < SLEN; i += 1024) { float e = __expf(sc[i] - m); sc[i] = e; ss += e; }
  red[tid] = ss; __syncthreads();
  for (int st = 512; st; st >>= 1) { if (tid < st) red[tid] += red[tid + st]; __syncthreads(); }
  float inv = 1.f / red[0]; __syncthreads();
  for (int i = tid; i < SLEN; i += 1024) sc[i] *= inv;
  __syncthreads();
  if (tid < 512) {
    float a = 0.f;
    for (int t = 0; t < SLEN; ++t) a += sc[t] * lstm_out[(size_t)t * 512 + tid];
    ao[tid] = a;
  }
  __syncthreads();
  if (tid < 8) {
    float b = 0.f;
    if (tid < KTAG) {
      float wa = 0.f;
      for (int h = 0; h < 512; ++h) wa += ao[h] * W_tag[tid * 512 + h];
      b = attn_w[0] * wa + b_tag[tid];
    }
    bias2[tid] = b;
  }
}

// ---------------- emissions: feats[t][k] (padded to 8 cols) ----------------

__global__ void feats_k(const float* __restrict__ lstm_out, const float* __restrict__ W_tag,
                        const float* __restrict__ bias2, const float* __restrict__ lf_prob,
                        const int* __restrict__ sent, float* __restrict__ feats) {
  int idx = blockIdx.x * blockDim.x + threadIdx.x;
  if (idx >= SLEN * 8) return;
  int t = idx >> 3, k = idx & 7;
  float v = NEGV;
  if (k < KTAG) {
    const float* lo = lstm_out + (size_t)t * 512;
    const float* w = W_tag + k * 512;
    float d = 0.f;
    for (int h = 0; h < 512; ++h) d += lo[h] * w[h];
    v = d + bias2[k] + lf_prob[(size_t)sent[t] * KTAG + k];
  }
  feats[idx] = v;
}

// ---------------- Viterbi (single wave, LDS backpointers, 8-deep feat prefetch) ----

__global__ __launch_bounds__(32) void viterbi_k(const float* __restrict__ feats,
                                                const float* __restrict__ trans,
                                                float* __restrict__ out) {
  __shared__ float tr[40];
  __shared__ float fv[8];
  __shared__ float term[8];
  __shared__ unsigned char bps[SLEN * 8];
  int l = threadIdx.x;
  if (l < 36) tr[l] = trans[l];
  if (l < 8) { fv[l] = (l == 0) ? 0.f : NEGV; term[l] = -3.0e38f; }
  __syncthreads();
  bool act = (l < KTAG);
  float pf[8];
#pragma unroll
  for (int i = 0; i < 8; ++i) pf[i] = act ? feats[i * 8 + l] : 0.f;
  for (int tb = 0; tb < SLEN; tb += 8) {
#pragma unroll
    for (int u = 0; u < 8; ++u) {
      int t = tb + u;
      if (act) {
        float best = -3.0e38f; int bp = 0;
#pragma unroll
        for (int p = 0; p < KTAG; ++p) {
          float s2 = fv[p] + tr[l * KTAG + p];
          if (s2 > best) { best = s2; bp = p; }
        }
        float nf = best + pf[u];
        int nt2 = t + 8;
        pf[u] = (nt2 < SLEN) ? feats[nt2 * 8 + l] : 0.f;
        fv[l] = nf;
        bps[t * 8 + l] = (unsigned char)bp;
      }
    }
  }
  if (act) term[l] = fv[l] + tr[1 * KTAG + l];   // STOP row = transitions[1]
  if (l == 0) {
    int bi = 0; float bs = term[0];
    for (int k2 = 1; k2 < KTAG; ++k2) if (term[k2] > bs) { bs = term[k2]; bi = k2; }
    out[SLEN] = bs;
    int tag = bi;
    for (int t = SLEN - 1; t >= 0; --t) { out[t] = (float)tag; tag = bps[t * 8 + tag]; }
  }
}

// ---------------- launch ----------------

extern "C" void kernel_launch(void* const* d_in, const int* in_sizes, int n_in,
                              void* d_out, int out_size, void* d_ws, size_t ws_size,
                              hipStream_t stream) {
  const int*   sent    = (const int*)d_in[0];
  const float* emb     = (const float*)d_in[1];
  const float* lf_prob = (const float*)d_in[2];
  const float* W_ih_f  = (const float*)d_in[3];
  const float* W_hh_f  = (const float*)d_in[4];
  const float* b_f     = (const float*)d_in[5];
  const float* W_ih_b  = (const float*)d_in[6];
  const float* W_hh_b  = (const float*)d_in[7];
  const float* b_b     = (const float*)d_in[8];
  const float* h0      = (const float*)d_in[9];
  const float* c0      = (const float*)d_in[10];
  const float* attn_w  = (const float*)d_in[11];
  const float* W_tag   = (const float*)d_in[12];
  const float* b_tag   = (const float*)d_in[13];
  const float* trans   = (const float*)d_in[14];

  char* ws = (char*)d_ws;
  unsigned short* Ap  = (unsigned short*)(ws);              // 2,621,440 B
  unsigned short* Bp  = (unsigned short*)(ws + 2621440);    // 1,310,720 B
  unsigned short* Bhh = (unsigned short*)(ws + 3932160);    // 1,048,576 B
  float* xW        = (float*)(ws + 4980736);                // 33,554,432 B
  float* lstm_out  = (float*)(ws + 38535168);               // 8,388,608 B
  float* final_vec = (float*)(ws + 46923776);               // 2,048 B
  float* bias2     = (float*)(ws + 46925824);               // 256 B
  float* feats     = (float*)(ws + 46926080);               // 131,072 B
  float* outp      = (float*)d_out;

  pack_a  <<<320, 256, 0, stream>>>(sent, emb, Ap);
  pack_bih<<<160, 256, 0, stream>>>(W_ih_f, W_ih_b, Bp);
  pack_bhh<<<128, 256, 0, stream>>>(W_hh_f, W_hh_b, Bhh);
  gemm_xw <<<1024, 256, 0, stream>>>(Ap, Bp, b_f, b_b, xW);
  lstm_k  <<<2, 1024, 0, stream>>>(Bhh, xW, h0, c0, lstm_out, final_vec);
  attn_k  <<<1, 1024, 0, stream>>>(lstm_out, final_vec, attn_w, W_tag, b_tag, bias2);
  feats_k <<<128, 256, 0, stream>>>(lstm_out, W_tag, bias2, lf_prob, sent, feats);
  viterbi_k<<<1, 32, 0, stream>>>(feats, trans, outp);
}
